// FraudGNN_15513421873288
// MI455X (gfx1250) — compile-verified, matrix operand order checked by reading, not run on hardware
//
#include <hip/hip_runtime.h>
#include <hip/hip_bf16.h>
#include <stdint.h>

// ---------------- types ----------------
typedef __attribute__((ext_vector_type(16))) __bf16        v16bf;
typedef __attribute__((ext_vector_type(8)))  float         v8f;
typedef __attribute__((ext_vector_type(4)))  unsigned int  v4u;

union FragU { v4u u[2]; v16bf v; };   // 32 bytes = 16 bf16

#define N_NODES 50000
#define N_EDGES 800000
#define FDIM    128
#define BN_EPS  1e-5f

static __device__ __forceinline__ __bf16 f2bf(float f) { return (__bf16)f; }

// ---------------- conversion kernels ----------------
__global__ void k_f32_to_bf16(const float* __restrict__ in, __bf16* __restrict__ out, int n) {
    int i = blockIdx.x * blockDim.x + threadIdx.x;
    if (i < n) out[i] = f2bf(in[i]);
}

// W is (K x N) row-major fp32; write W^T as (N x K) row-major bf16 so B-fragments
// load contiguously (column-major B == row-major B^T).
__global__ void k_w_to_bf16_t(const float* __restrict__ W, __bf16* __restrict__ Wt, int K, int N) {
    int idx = blockIdx.x * blockDim.x + threadIdx.x;
    if (idx < K * N) {
        int k = idx / N, n = idx % N;
        Wt[n * K + k] = f2bf(W[idx]);
    }
}

// ---------------- degree / norm ----------------
__global__ void k_deg_init(float* __restrict__ deg) {
    int i = blockIdx.x * blockDim.x + threadIdx.x;
    if (i < N_NODES) deg[i] = 1.0f;  // self loop
}
__global__ void k_deg_count(const int* __restrict__ dst, float* __restrict__ deg) {
    int e = blockIdx.x * blockDim.x + threadIdx.x;
    if (e < N_EDGES) atomicAdd(&deg[dst[e]], 1.0f);
}
__global__ void k_deg_rsqrt(float* __restrict__ deg) {
    int i = blockIdx.x * blockDim.x + threadIdx.x;
    if (i < N_NODES) deg[i] = rsqrtf(deg[i]);   // in place: deg -> dinv
}

// ---------------- WMMA GEMM: C = A(MxK,bf16) * B(KxN,bf16 given as B^T NxK) ----------------
// One wave per 16x16 output tile; block = (N/16) waves covering 16 rows x N cols.
// A-fragment lane layout per CDNA5 ISA (16-bit A 16x32): lane L holds row M=L&15;
// lanes 0-15 carry K = {k0+0..7, k0+16..23}, lanes 16-31 carry K = {k0+8..15, k0+24..31}.
// Both halves load as two 16B vectors since K runs contiguously within each group of 8.
__global__ void k_wmma_gemm(const __bf16* __restrict__ A,
                            const __bf16* __restrict__ Bt,
                            const float*  __restrict__ bias,   // may be null
                            float*        __restrict__ C,
                            __bf16*       __restrict__ Cb,     // optional bf16 shadow, may be null
                            int N, int K, int relu)
{
    const int lane = threadIdx.x & 31;
    const int half = lane >> 4;      // 0: low-K half, 1: high-K half
    const int ln   = lane & 15;
    const int row0 = blockIdx.x * 16;
    const int col0 = (threadIdx.x >> 5) * 16;

    const __bf16* arow = A  + (size_t)(row0 + ln) * K;
    const __bf16* brow = Bt + (size_t)(col0 + ln) * K;

    v8f acc = {};
#pragma unroll
    for (int k0 = 0; k0 < 128; k0 += 32) {
        FragU a, b;
        a.u[0] = *(const v4u*)(arow + k0 + 8 * half);        // K = k0+8h .. +8
        a.u[1] = *(const v4u*)(arow + k0 + 16 + 8 * half);   // K = k0+16+8h .. +8
        b.u[0] = *(const v4u*)(brow + k0 + 8 * half);
        b.u[1] = *(const v4u*)(brow + k0 + 16 + 8 * half);
        acc = __builtin_amdgcn_wmma_f32_16x16x32_bf16(
                  false, a.v, false, b.v, (short)0, acc, false, false);
    }

    const float bval = bias ? bias[col0 + ln] : 0.0f;
#pragma unroll
    for (int r = 0; r < 8; ++r) {
        int m = row0 + r + 8 * half;          // C/D layout: lanes16-31 hold M=8..15
        float v = acc[r] + bval;
        if (relu) v = fmaxf(v, 0.0f);
        size_t o = (size_t)m * N + col0 + ln;
        C[o] = v;
        if (Cb) Cb[o] = f2bf(v);
    }
}

// ---------------- GCN aggregation ----------------
// agg[i][f] = h[i][f] * dinv[i]^2 + bias[f]   (self message + bias, also zero-inits accumulator)
__global__ void k_agg_init(const float* __restrict__ h, const float* __restrict__ dinv,
                           const float* __restrict__ bias, float* __restrict__ agg) {
    int idx = blockIdx.x * blockDim.x + threadIdx.x;
    if (idx < N_NODES * FDIM) {
        int i = idx >> 7, f = idx & 127;
        float d = dinv[i];
        agg[idx] = h[idx] * d * d + bias[f];
    }
}

// one wave per edge; each lane handles 4 contiguous features (float4 gather, 4 atomics)
__global__ void k_edge_scatter(const int* __restrict__ src, const int* __restrict__ dst,
                               const float* __restrict__ dinv,
                               const float* __restrict__ h, float* __restrict__ agg) {
    int t = blockIdx.x * blockDim.x + threadIdx.x;
    int e = t >> 5, lane = t & 31;
    if (e >= N_EDGES) return;
    int s = src[e], d = dst[e];
    float coef = dinv[s] * dinv[d];
    const float4* hp = (const float4*)(h + (size_t)s * FDIM);
    float4 v = hp[lane];
    float* ap = agg + (size_t)d * FDIM + lane * 4;
    atomicAdd(ap + 0, v.x * coef);
    atomicAdd(ap + 1, v.y * coef);
    atomicAdd(ap + 2, v.z * coef);
    atomicAdd(ap + 3, v.w * coef);
}

// ---------------- batch norm ----------------
__global__ void k_stats_zero(float* __restrict__ stats) {
    int i = threadIdx.x;
    if (i < 2 * FDIM) stats[i] = 0.0f;
}
__global__ void k_stats(const float* __restrict__ agg, float* __restrict__ stats) {
    int f = threadIdx.x;           // 128 threads
    float s = 0.0f, sq = 0.0f;
    for (int i = blockIdx.x; i < N_NODES; i += gridDim.x) {
        float v = agg[(size_t)i * FDIM + f];
        s += v; sq += v * v;
    }
    atomicAdd(&stats[f], s);
    atomicAdd(&stats[FDIM + f], sq);
}
// relu(bn(agg)) -> hout (f32, for edge gather) and hbf (bf16, feeds next WMMA GEMM)
__global__ void k_bn_relu(const float* __restrict__ agg, const float* __restrict__ stats,
                          const float* __restrict__ g, const float* __restrict__ beta,
                          float* __restrict__ hout, __bf16* __restrict__ hbf) {
    int idx = blockIdx.x * blockDim.x + threadIdx.x;
    if (idx < N_NODES * FDIM) {
        int f = idx & 127;
        const float invN = 1.0f / (float)N_NODES;
        float m   = stats[f] * invN;
        float var = stats[FDIM + f] * invN - m * m;
        float v = (agg[idx] - m) * rsqrtf(var + BN_EPS) * g[f] + beta[f];
        v = fmaxf(v, 0.0f);
        hout[idx] = v;
        hbf[idx]  = f2bf(v);
    }
}

// ---------------- final tiny FC (64 -> 2) ----------------
__global__ void k_fc2(const float* __restrict__ hid, const float* __restrict__ fw2,
                      const float* __restrict__ fb2, float* __restrict__ out) {
    int i = blockIdx.x * blockDim.x + threadIdx.x;
    if (i >= N_NODES) return;
    const float4* hr = (const float4*)(hid + (size_t)i * 64);
    float a0 = fb2[0], a1 = fb2[1];
#pragma unroll
    for (int q = 0; q < 16; ++q) {
        float4 hv = hr[q];
        int k = q * 4;
        a0 += hv.x * fw2[(k+0)*2] + hv.y * fw2[(k+1)*2] + hv.z * fw2[(k+2)*2] + hv.w * fw2[(k+3)*2];
        a1 += hv.x * fw2[(k+0)*2+1] + hv.y * fw2[(k+1)*2+1] + hv.z * fw2[(k+2)*2+1] + hv.w * fw2[(k+3)*2+1];
    }
    out[i * 2 + 0] = a0;
    out[i * 2 + 1] = a1;
}

// ---------------- host side ----------------
static inline size_t alignup(size_t x) { return (x + 255) & ~(size_t)255; }

extern "C" void kernel_launch(void* const* d_in, const int* in_sizes, int n_in,
                              void* d_out, int out_size, void* d_ws, size_t ws_size,
                              hipStream_t stream) {
    (void)in_sizes; (void)n_in; (void)out_size; (void)ws_size;

    const float* x    = (const float*)d_in[0];
    const int*   src  = (const int*)  d_in[1];
    const int*   dst  = (const int*)  d_in[2];
    const float* W1   = (const float*)d_in[3];
    const float* b1   = (const float*)d_in[4];
    const float* g1   = (const float*)d_in[5];
    const float* bt1  = (const float*)d_in[6];
    const float* W2   = (const float*)d_in[7];
    const float* b2   = (const float*)d_in[8];
    const float* g2   = (const float*)d_in[9];
    const float* bt2  = (const float*)d_in[10];
    const float* fw1  = (const float*)d_in[11];
    const float* fb1  = (const float*)d_in[12];
    const float* fw2  = (const float*)d_in[13];
    const float* fb2  = (const float*)d_in[14];
    float* out = (float*)d_out;

    // workspace layout
    char* ws = (char*)d_ws;
    size_t o = 0;
    float*  dinv  = (float*) (ws + o); o = alignup(o + (size_t)N_NODES * 4);
    __bf16* xb    = (__bf16*)(ws + o); o = alignup(o + (size_t)N_NODES * FDIM * 2);
    __bf16* Wb    = (__bf16*)(ws + o); o = alignup(o + (size_t)FDIM * FDIM * 2);
    float*  h     = (float*) (ws + o); o = alignup(o + (size_t)N_NODES * FDIM * 4);
    float*  agg   = (float*) (ws + o); o = alignup(o + (size_t)N_NODES * FDIM * 4);
    float*  stats = (float*) (ws + o); o = alignup(o + 2 * FDIM * 4);
    float*  hid   = agg;  // reuse: agg is dead once layer-2 BN is applied

    const int NF = N_NODES * FDIM;

    // degree -> dinv
    k_deg_init <<<(N_NODES + 255) / 256, 256, 0, stream>>>(dinv);
    k_deg_count<<<(N_EDGES + 255) / 256, 256, 0, stream>>>(dst, dinv);
    k_deg_rsqrt<<<(N_NODES + 255) / 256, 256, 0, stream>>>(dinv);

    // x -> bf16
    k_f32_to_bf16<<<(NF + 255) / 256, 256, 0, stream>>>(x, xb, NF);

    // ---- layer 1 ----
    k_w_to_bf16_t<<<(FDIM * FDIM + 255) / 256, 256, 0, stream>>>(W1, Wb, FDIM, FDIM);
    k_wmma_gemm<<<N_NODES / 16, 256, 0, stream>>>(xb, Wb, nullptr, h, nullptr, FDIM, FDIM, 0);
    k_agg_init<<<(NF + 255) / 256, 256, 0, stream>>>(h, dinv, b1, agg);
    k_edge_scatter<<<(N_EDGES * 32 + 255) / 256, 256, 0, stream>>>(src, dst, dinv, h, agg);
    k_stats_zero<<<1, 256, 0, stream>>>(stats);
    k_stats<<<512, FDIM, 0, stream>>>(agg, stats);
    k_bn_relu<<<(NF + 255) / 256, 256, 0, stream>>>(agg, stats, g1, bt1, h, xb);

    // ---- layer 2 ----
    k_w_to_bf16_t<<<(FDIM * FDIM + 255) / 256, 256, 0, stream>>>(W2, Wb, FDIM, FDIM);
    k_wmma_gemm<<<N_NODES / 16, 256, 0, stream>>>(xb, Wb, nullptr, h, nullptr, FDIM, FDIM, 0);
    k_agg_init<<<(NF + 255) / 256, 256, 0, stream>>>(h, dinv, b2, agg);
    k_edge_scatter<<<(N_EDGES * 32 + 255) / 256, 256, 0, stream>>>(src, dst, dinv, h, agg);
    k_stats_zero<<<1, 256, 0, stream>>>(stats);
    k_stats<<<512, FDIM, 0, stream>>>(agg, stats);
    k_bn_relu<<<(NF + 255) / 256, 256, 0, stream>>>(agg, stats, g2, bt2, h, xb);

    // ---- MLP head: relu(xb @ fw1 + fb1) via WMMA (N=64), then 64->2 ----
    k_w_to_bf16_t<<<(FDIM * 64 + 255) / 256, 256, 0, stream>>>(fw1, Wb, FDIM, 64);
    k_wmma_gemm<<<N_NODES / 16, 128, 0, stream>>>(xb, Wb, fb1, hid, nullptr, 64, FDIM, 1);
    k_fc2<<<(N_NODES + 255) / 256, 256, 0, stream>>>(hid, fw2, fb2, out);
}
